// PatchSelfAttentionBlock_28518582845618
// MI455X (gfx1250) — matrix operand order repaired and checked by reading, not compiled, not run on hardware
//
#include <hip/hip_runtime.h>
#include <math.h>

// ---------------------------------------------------------------------------
// Types for CDNA5 WMMA (wave32)
// ---------------------------------------------------------------------------
typedef __bf16 bf16;
typedef __attribute__((ext_vector_type(16))) __bf16 v16bf;
typedef __attribute__((ext_vector_type(8)))  __bf16 v8bf;
typedef __attribute__((ext_vector_type(8)))  float  v8f;

#define CAT16(lo, hi) __builtin_shufflevector((lo), (hi), 0,1,2,3,4,5,6,7,8,9,10,11,12,13,14,15)

// Problem constants
#define PSZ     16
#define NHEADS  8
#define CIN     256
#define HIMG    512
#define WIMG    512
#define BATCH   2
#define HP      32            // patches per side
#define NPATCH  1024          // HP*HP
#define KDIM    65536         // CIN*PSZ*PSZ
#define OC      512           // NHEADS*64
#define OC3     1536          // stacked q,k,v rows

// GEMM tiling
#define BK 64
#define ST 72                 // padded LDS row stride (bf16 elems)

static __device__ __forceinline__ unsigned short f2bfu(float f) {
  unsigned u = __builtin_bit_cast(unsigned, f);
  unsigned r = u + 0x7FFFu + ((u >> 16) & 1u);
  return (unsigned short)(r >> 16);
}
static __device__ __forceinline__ bf16 f2bf(float f) {
  unsigned short h = f2bfu(f);
  return __builtin_bit_cast(bf16, h);
}

// gfx1250 async global->LDS copy (16B per lane), tracked by ASYNCcnt.
// LDS address = low 32 bits of the generic pointer (LDS aperture rule).
static __device__ __forceinline__ void async_ld_b128(void* lds, const void* gptr) {
  unsigned loff = (unsigned)(size_t)lds;
  asm volatile("global_load_async_to_lds_b128 %0, %1, off"
               :: "v"(loff), "v"(gptr) : "memory");
}
static __device__ __forceinline__ void wait_async0() {
  asm volatile("s_wait_asynccnt 0x0" ::: "memory");
}

// ---------------------------------------------------------------------------
// Prep: x [B,C,H,W] fp32  ->  xb [B][n=patch][k=c*256+p*16+q] bf16
// ---------------------------------------------------------------------------
__global__ __launch_bounds__(256) void prep_x_kernel(const float* __restrict__ x,
                                                     bf16* __restrict__ xb) {
  size_t idx = ((size_t)blockIdx.x * 256u + threadIdx.x) * 4u;
  size_t t = idx;
  int xx = (int)(t % WIMG); t /= WIMG;
  int y  = (int)(t % HIMG); t /= HIMG;
  int c  = (int)(t % CIN);  t /= CIN;
  int b  = (int)t;
  float4 v = *(const float4*)(x + idx);
  int n = (y >> 4) * HP + (xx >> 4);
  int k = c * 256 + (y & 15) * 16 + (xx & 15);
  bf16* dst = xb + ((size_t)(b * NPATCH + n)) * KDIM + k;
  uint2 o;
  o.x = (unsigned)f2bfu(v.x) | ((unsigned)f2bfu(v.y) << 16);
  o.y = (unsigned)f2bfu(v.z) | ((unsigned)f2bfu(v.w) << 16);
  *(uint2*)dst = o;
}

// Prep: stack wq,wk,wv into wb [1536][65536] bf16 (K contiguous)
__global__ __launch_bounds__(256) void prep_wqkv_kernel(const float* __restrict__ wq,
                                                        const float* __restrict__ wk,
                                                        const float* __restrict__ wv,
                                                        bf16* __restrict__ wb) {
  size_t idx = ((size_t)blockIdx.x * 256u + threadIdx.x) * 4u;
  size_t row = idx / KDIM, col = idx % KDIM;
  const float* src;
  if (row < 512)       src = wq + row * (size_t)KDIM + col;
  else if (row < 1024) src = wk + (row - 512) * (size_t)KDIM + col;
  else                 src = wv + (row - 1024) * (size_t)KDIM + col;
  float4 v = *(const float4*)src;
  uint2 o;
  o.x = (unsigned)f2bfu(v.x) | ((unsigned)f2bfu(v.y) << 16);
  o.y = (unsigned)f2bfu(v.z) | ((unsigned)f2bfu(v.w) << 16);
  *(uint2*)(wb + idx) = o;
}

// Prep: transpose wo [512][65536] -> wob [m=65536][o=512] bf16
__global__ __launch_bounds__(256) void prep_wo_kernel(const float* __restrict__ wo,
                                                      bf16* __restrict__ wob) {
  size_t idx = ((size_t)blockIdx.x * 256u + threadIdx.x) * 4u;
  size_t o = idx / KDIM, m = idx % KDIM;
  float4 v = *(const float4*)(wo + idx);
  wob[(m + 0) * 512 + o] = f2bf(v.x);
  wob[(m + 1) * 512 + o] = f2bf(v.y);
  wob[(m + 2) * 512 + o] = f2bf(v.z);
  wob[(m + 3) * 512 + o] = f2bf(v.w);
}

// Prep: 2D sinusoidal PE, pe[ch][n] fp32, ch<512, n = h*32+w
__global__ __launch_bounds__(256) void prep_pe_kernel(float* __restrict__ pe) {
  int idx = blockIdx.x * 256 + threadIdx.x;   // 512*1024
  int ch = idx >> 10, n = idx & 1023;
  int hh = n >> 5, ww = n & 31;
  int cc, pos;
  if (ch < 256) { cc = ch;       pos = hh; }
  else          { cc = ch - 256; pos = ww; }
  int m = cc >> 1;
  float freq = expf((float)(2 * m) * (-logf(10000.f) / 256.f));
  float ang = (float)pos * freq;
  pe[idx] = (cc & 1) ? cosf(ang) : sinf(ang);
}

// ---------------------------------------------------------------------------
// Fused q/k/v projection GEMM: [1536 x 65536] x [65536 x 1024] per batch.
// 256 threads (8 waves), 128x128 tile, BK=64, bf16 WMMA, fp32 accum.
// Double-buffered ASYNC global->LDS staging overlapped with WMMA.
// Epilogue adds PE, converts bf16, scatters to qs/ks [bh][n][d], vs [bh][d][n].
// ---------------------------------------------------------------------------
__global__ __launch_bounds__(256) void gemm_qkv_kernel(
    const bf16* __restrict__ wb, const bf16* __restrict__ xb,
    const float* __restrict__ pe,
    bf16* __restrict__ qs, bf16* __restrict__ ksm, bf16* __restrict__ vs) {
  __shared__ bf16 As[2][128 * ST];
  __shared__ bf16 Bs[2][128 * ST];
  int tid = threadIdx.x;
  int blk = blockIdx.x;                 // 96 = 12 mblk x 8 nblk
  int b   = blockIdx.y;
  int mblk = blk >> 3, nblk = blk & 7;
  size_t mbase = (size_t)mblk * 128, nbase = (size_t)nblk * 128;
  int wave = tid >> 5, lane = tid & 31;
  int hl = lane >> 4, r = lane & 15;
  int mw = wave & 3, nw = wave >> 2;    // wave owns 2 m-subtiles x 4 n-subtiles

  v8f acc[2][4];
#pragma unroll
  for (int i = 0; i < 2; ++i)
#pragma unroll
    for (int j = 0; j < 4; ++j)
#pragma unroll
      for (int e = 0; e < 8; ++e) acc[i][j][e] = 0.f;

  const bf16* Ag = wb + mbase * KDIM;
  const bf16* Bg = xb + ((size_t)b * NPATCH + nbase) * KDIM;

  int row0 = tid >> 3;                  // staging: 4 chunks of 8 bf16 per matrix
  int coff = (tid & 7) * 8;

  auto issue = [&](int buf, int k0) {
#pragma unroll
    for (int j = 0; j < 4; ++j) {
      int row = row0 + 32 * j;
      async_ld_b128(&As[buf][row * ST + coff], Ag + (size_t)row * KDIM + k0 + coff);
      async_ld_b128(&Bs[buf][row * ST + coff], Bg + (size_t)row * KDIM + k0 + coff);
    }
  };

  issue(0, 0);
  wait_async0();
  __syncthreads();

  int cur = 0;
#pragma unroll 1
  for (int k0 = 0; k0 < KDIM; k0 += BK) {
    if (k0 + BK < KDIM) issue(cur ^ 1, k0 + BK);
    const bf16* Ab = &As[cur][0];
    const bf16* Bb = &Bs[cur][0];
#pragma unroll
    for (int ks = 0; ks < 2; ++ks) {
      int kk = ks * 32;
      v16bf af[2], bfv[4];
#pragma unroll
      for (int im = 0; im < 2; ++im) {
        const bf16* p = Ab + ((2 * mw + im) * 16 + r) * ST + kk;
        v8bf lo = *(const v8bf*)(p + hl * 8);
        v8bf hi = *(const v8bf*)(p + 16 + hl * 8);
        af[im] = CAT16(lo, hi);
      }
#pragma unroll
      for (int in = 0; in < 4; ++in) {
        const bf16* p = Bb + ((4 * nw + in) * 16 + r) * ST + kk;
        v8bf lo = *(const v8bf*)(p + hl * 8);
        v8bf hi = *(const v8bf*)(p + 16 + hl * 8);
        bfv[in] = CAT16(lo, hi);
      }
#pragma unroll
      for (int im = 0; im < 2; ++im)
#pragma unroll
        for (int in = 0; in < 4; ++in)
          acc[im][in] = __builtin_amdgcn_wmma_f32_16x16x32_bf16(
              false, af[im], false, bfv[in], (short)0, acc[im][in], false, false);
    }
    wait_async0();
    __syncthreads();
    cur ^= 1;
  }

#pragma unroll
  for (int im = 0; im < 2; ++im) {
    int mt = (2 * mw + im) * 16;
#pragma unroll
    for (int in = 0; in < 4; ++in) {
      int nt = (4 * nw + in) * 16;
      int n = (int)nbase + nt + r;
#pragma unroll
      for (int i = 0; i < 8; ++i) {
        int o = (int)mbase + mt + i + 8 * hl;
        int oo = o & 511;
        float val = acc[im][in][i] + pe[(size_t)oo * NPATCH + n];
        int proj = o >> 9;
        int head = oo >> 6, d = oo & 63;
        int bh = b * NHEADS + head;
        bf16 bv = f2bf(val);
        if (proj == 0)      qs [((size_t)bh * NPATCH + n) * 64 + d] = bv;
        else if (proj == 1) ksm[((size_t)bh * NPATCH + n) * 64 + d] = bv;
        else                vs [((size_t)bh * 64 + d) * NPATCH + n] = bv;
      }
    }
  }
}

// ---------------------------------------------------------------------------
// Attention: one block per (bh, 16-row block). Scores via WMMA into 64 KB LDS,
// wave32 softmax (shfl reductions), attn*V via WMMA with on-the-fly f32->bf16.
// Writes outb [b][n][head*64+dv] bf16 (K-contiguous for the output GEMM).
// ---------------------------------------------------------------------------
__global__ __launch_bounds__(256) void attn_kernel(
    const bf16* __restrict__ qs, const bf16* __restrict__ ksm,
    const bf16* __restrict__ vs, bf16* __restrict__ outb) {
  __shared__ float Sc[16 * 1024];       // 64 KB
  int tid = threadIdx.x, wave = tid >> 5, lane = tid & 31;
  int hl = lane >> 4, r = lane & 15;
  int rowblk = blockIdx.x;              // 64 blocks of 16 rows
  int bh = blockIdx.y;                  // 16 = B*NHEADS
  int b = bh >> 3, head = bh & 7;
  const float scale = 0.125f;           // 1/sqrt(64)

  // ---- Phase 1: scores = Q K^T * scale ----
  v16bf af[2];
  {
    const bf16* qrow = qs + ((size_t)bh * NPATCH + (size_t)rowblk * 16 + r) * 64;
#pragma unroll
    for (int ks = 0; ks < 2; ++ks) {
      v8bf lo = *(const v8bf*)(qrow + ks * 32 + hl * 8);
      v8bf hi = *(const v8bf*)(qrow + ks * 32 + 16 + hl * 8);
      af[ks] = CAT16(lo, hi);
    }
  }
#pragma unroll 1
  for (int t = 0; t < 8; ++t) {
    int ns = wave * 8 + t;              // 64 column subtiles, 8 per wave
    const bf16* kcol = ksm + ((size_t)bh * NPATCH + (size_t)ns * 16 + r) * 64;
    v8f c;
#pragma unroll
    for (int e = 0; e < 8; ++e) c[e] = 0.f;
#pragma unroll
    for (int ks = 0; ks < 2; ++ks) {
      v8bf lo = *(const v8bf*)(kcol + ks * 32 + hl * 8);
      v8bf hi = *(const v8bf*)(kcol + ks * 32 + 16 + hl * 8);
      v16bf bfv = CAT16(lo, hi);
      c = __builtin_amdgcn_wmma_f32_16x16x32_bf16(false, af[ks], false, bfv,
                                                  (short)0, c, false, false);
    }
#pragma unroll
    for (int i = 0; i < 8; ++i)
      Sc[(i + 8 * hl) * 1024 + ns * 16 + r] = c[i] * scale;
  }
  __syncthreads();

  // ---- Phase 2: row softmax (2 rows per wave) ----
#pragma unroll 1
  for (int rr = 0; rr < 2; ++rr) {
    int row = wave * 2 + rr;
    float* Sr = &Sc[row * 1024];
    float mx = -3.4e38f;
    for (int c = lane; c < 1024; c += 32) mx = fmaxf(mx, Sr[c]);
#pragma unroll
    for (int s = 16; s; s >>= 1) mx = fmaxf(mx, __shfl_xor(mx, s, 32));
    float sum = 0.f;
    for (int c = lane; c < 1024; c += 32) {
      float e = __expf(Sr[c] - mx);
      Sr[c] = e;
      sum += e;
    }
#pragma unroll
    for (int s = 16; s; s >>= 1) sum += __shfl_xor(sum, s, 32);
    float inv = 1.f / sum;
    for (int c = lane; c < 1024; c += 32) Sr[c] *= inv;
  }
  __syncthreads();

  // ---- Phase 3: out = attn * V (waves 0..3, one 16x16 dv-subtile each) ----
  if (wave < 4) {
    v8f acc;
#pragma unroll
    for (int e = 0; e < 8; ++e) acc[e] = 0.f;
    const bf16* vcol = vs + ((size_t)bh * 64 + (size_t)wave * 16 + r) * NPATCH;
    const float* Sr = &Sc[r * 1024];
#pragma unroll 1
    for (int k0 = 0; k0 < 1024; k0 += 32) {
      v8f lo = *(const v8f*)(Sr + k0 + hl * 8);
      v8f hi = *(const v8f*)(Sr + k0 + 16 + hl * 8);
      v16bf a;
#pragma unroll
      for (int e = 0; e < 8; ++e) { a[e] = f2bf(lo[e]); a[e + 8] = f2bf(hi[e]); }
      v8bf blo = *(const v8bf*)(vcol + k0 + hl * 8);
      v8bf bhi = *(const v8bf*)(vcol + k0 + 16 + hl * 8);
      v16bf bfv = CAT16(blo, bhi);
      acc = __builtin_amdgcn_wmma_f32_16x16x32_bf16(false, a, false, bfv,
                                                    (short)0, acc, false, false);
    }
#pragma unroll
    for (int i = 0; i < 8; ++i) {
      int nrow = rowblk * 16 + i + 8 * hl;
      int dv = wave * 16 + r;
      outb[((size_t)b * NPATCH + nrow) * 512 + head * 64 + dv] = f2bf(acc[i]);
    }
  }
}

// ---------------------------------------------------------------------------
// Output projection GEMM: y[m=cpq][n] = sum_o wob[m][o] * outb[b][n][o] + bo
// [65536 x 512] x [512 x 1024] per batch; async double-buffered staging;
// scatter back to pixels.
// ---------------------------------------------------------------------------
__global__ __launch_bounds__(256) void gemm_out_kernel(
    const bf16* __restrict__ wob, const bf16* __restrict__ outb,
    const float* __restrict__ bo, float* __restrict__ y) {
  __shared__ bf16 As[2][128 * ST];
  __shared__ bf16 Bs[2][128 * ST];
  int tid = threadIdx.x;
  int blk = blockIdx.x;                 // 4096 = 512 mblk x 8 nblk
  int b   = blockIdx.y;
  int mblk = blk >> 3, nblk = blk & 7;
  size_t mbase = (size_t)mblk * 128, nbase = (size_t)nblk * 128;
  int wave = tid >> 5, lane = tid & 31;
  int hl = lane >> 4, r = lane & 15;
  int mw = wave & 3, nw = wave >> 2;

  v8f acc[2][4];
#pragma unroll
  for (int i = 0; i < 2; ++i)
#pragma unroll
    for (int j = 0; j < 4; ++j)
#pragma unroll
      for (int e = 0; e < 8; ++e) acc[i][j][e] = 0.f;

  const bf16* Ag = wob + mbase * 512;
  const bf16* Bg = outb + ((size_t)b * NPATCH + nbase) * 512;

  int row0 = tid >> 3;
  int coff = (tid & 7) * 8;

  auto issue = [&](int buf, int k0) {
#pragma unroll
    for (int j = 0; j < 4; ++j) {
      int row = row0 + 32 * j;
      async_ld_b128(&As[buf][row * ST + coff], Ag + (size_t)row * 512 + k0 + coff);
      async_ld_b128(&Bs[buf][row * ST + coff], Bg + (size_t)row * 512 + k0 + coff);
    }
  };

  issue(0, 0);
  wait_async0();
  __syncthreads();

  int cur = 0;
#pragma unroll 1
  for (int k0 = 0; k0 < 512; k0 += BK) {
    if (k0 + BK < 512) issue(cur ^ 1, k0 + BK);
    const bf16* Ab = &As[cur][0];
    const bf16* Bb = &Bs[cur][0];
#pragma unroll
    for (int ks = 0; ks < 2; ++ks) {
      int kk = ks * 32;
      v16bf af[2], bfv[4];
#pragma unroll
      for (int im = 0; im < 2; ++im) {
        const bf16* p = Ab + ((2 * mw + im) * 16 + r) * ST + kk;
        v8bf lo = *(const v8bf*)(p + hl * 8);
        v8bf hi = *(const v8bf*)(p + 16 + hl * 8);
        af[im] = CAT16(lo, hi);
      }
#pragma unroll
      for (int in = 0; in < 4; ++in) {
        const bf16* p = Bb + ((4 * nw + in) * 16 + r) * ST + kk;
        v8bf lo = *(const v8bf*)(p + hl * 8);
        v8bf hi = *(const v8bf*)(p + 16 + hl * 8);
        bfv[in] = CAT16(lo, hi);
      }
#pragma unroll
      for (int im = 0; im < 2; ++im)
#pragma unroll
        for (int in = 0; in < 4; ++in)
          acc[im][in] = __builtin_amdgcn_wmma_f32_16x16x32_bf16(
              false, af[im], false, bfv[in], (short)0, acc[im][in], false, false);
    }
    wait_async0();
    __syncthreads();
    cur ^= 1;
  }

#pragma unroll
  for (int im = 0; im < 2; ++im) {
    int mt = (2 * mw + im) * 16;
#pragma unroll
    for (int in = 0; in < 4; ++in) {
      int nt = (4 * nw + in) * 16;
      int n = (int)nbase + nt + r;
      int ii = n >> 5, jj = n & 31;
#pragma unroll
      for (int i = 0; i < 8; ++i) {
        int m = (int)mbase + mt + i + 8 * hl;
        int c = m >> 8, pq = m & 255;
        int pp = pq >> 4, qq = pq & 15;
        float val = acc[im][in][i] + bo[c];
        y[(((size_t)(b * CIN + c) * HIMG) + ii * 16 + pp) * WIMG + jj * 16 + qq] = val;
      }
    }
  }
}

// ---------------------------------------------------------------------------
// Launch
// ---------------------------------------------------------------------------
extern "C" void kernel_launch(void* const* d_in, const int* in_sizes, int n_in,
                              void* d_out, int out_size, void* d_ws, size_t ws_size,
                              hipStream_t stream) {
  (void)in_sizes; (void)n_in; (void)out_size; (void)ws_size;
  const float* x  = (const float*)d_in[0];
  const float* wq = (const float*)d_in[1];
  const float* wk = (const float*)d_in[2];
  const float* wv = (const float*)d_in[3];
  const float* wo = (const float*)d_in[4];
  const float* bo = (const float*)d_in[5];
  float* y = (float*)d_out;

  char* ws = (char*)d_ws;
  bf16*  xb   = (bf16*)(ws + 0ull);            // 256 MB  x in bf16 patch-major
  bf16*  wb   = (bf16*)(ws + 268435456ull);    // 192 MB  stacked wq|wk|wv bf16
  bf16*  wob  = (bf16*)(ws + 469762048ull);    //  64 MB  wo transposed bf16
  float* pe   = (float*)(ws + 536870912ull);   //   2 MB  positional encoding
  bf16*  qs   = (bf16*)(ws + 538968064ull);    //   2 MB  Q [bh][n][d]
  bf16*  ksm  = (bf16*)(ws + 541065216ull);    //   2 MB  K [bh][n][d]
  bf16*  vs   = (bf16*)(ws + 543162368ull);    //   2 MB  V [bh][d][n]
  bf16*  outb = (bf16*)(ws + 545259520ull);    //   2 MB  attn out [b][n][o]

  prep_x_kernel   <<<131072, 256, 0, stream>>>(x, xb);
  prep_wqkv_kernel<<< 98304, 256, 0, stream>>>(wq, wk, wv, wb);
  prep_wo_kernel  <<< 32768, 256, 0, stream>>>(wo, wob);
  prep_pe_kernel  <<<  2048, 256, 0, stream>>>(pe);

  gemm_qkv_kernel <<<dim3(96, 2),   256, 0, stream>>>(wb, xb, pe, qs, ksm, vs);
  attn_kernel     <<<dim3(64, 16),  256, 0, stream>>>(qs, ksm, vs, outb);
  gemm_out_kernel <<<dim3(4096, 2), 256, 0, stream>>>(wob, outb, bo, y);
}
